// TransformerNet_2156073582617
// MI455X (gfx1250) — matrix operand order repaired and checked by reading, not compiled, not run on hardware
//
#include <hip/hip_runtime.h>
#include <cstdint>

// ---------------------------------------------------------------------------
// Graph TransformerConv (PyG-style) x2 + LN/ReLU + classifier head, MI455X.
//   GEMMs: bf16 v_wmma_f32_16x16x32_bf16, fp32 accumulate, X-tile staged once
//          in LDS and reused for Wq/Wk/Wv/Ws (saves 3x re-read of 51MB acts).
//   Edge phases: L2-resident gathers + native f32 atomics (max/add).
// ---------------------------------------------------------------------------

typedef __attribute__((ext_vector_type(16))) __bf16 v16bf;
typedef __attribute__((ext_vector_type(8)))  float  v8f;

#define HIDDIM   128
#define HEADS    4
#define NCLS_    40
#define NCLS_PAD 48
#define LSTRIDE  136   // LDS row stride in ushorts: 272B, 16B aligned, padded

union FragU { v16bf v; uint4 q[2]; };

__device__ __forceinline__ unsigned short f2bf(float f) {
  unsigned int u = __float_as_uint(f);
  unsigned int r = u + 0x7FFFu + ((u >> 16) & 1u);   // round-to-nearest-even
  return (unsigned short)(r >> 16);
}

__device__ __forceinline__ void atomicMaxF(float* addr, float v) {
  // monotonic-bits trick -> native global_atomic_max/min, no CAS loop
  if (v >= 0.0f) atomicMax((int*)addr, __float_as_int(v));
  else           atomicMin((unsigned int*)addr, __float_as_uint(v));
}

// ---------------- weight pre-transpose + bf16 conversion -------------------
// W: [128][128] row-major (k,n)  ->  Wt: [n][k] bf16
__global__ __launch_bounds__(256) void convt128_kernel(
    const float* __restrict__ W, unsigned short* __restrict__ Wt) {
  int idx = blockIdx.x * 256 + threadIdx.x;
  if (idx < 128 * 128) {
    int n = idx >> 7, k = idx & 127;
    Wt[idx] = f2bf(W[k * 128 + n]);
  }
}

// Wout: [128][40] -> Wt: [48][128] bf16, zero-padded cols 40..47
__global__ __launch_bounds__(256) void convtout_kernel(
    const float* __restrict__ W, unsigned short* __restrict__ Wt) {
  int idx = blockIdx.x * 256 + threadIdx.x;
  if (idx < NCLS_PAD * 128) {
    int n = idx >> 7, k = idx & 127;
    Wt[idx] = (n < NCLS_) ? f2bf(W[k * NCLS_ + n]) : (unsigned short)0;
  }
}

__global__ __launch_bounds__(256) void init_attn_kernel(
    float* __restrict__ amax, float* __restrict__ denom, int n) {
  int i = blockIdx.x * 256 + threadIdx.x;
  if (i < n) { amax[i] = -INFINITY; denom[i] = 0.0f; }
}

// ---------------- fused 4-way GEMM: Q,K,V,AGG(=skip) = X @ {Wq,Wk,Wv,Ws}+b --
// block = 256 threads (8 waves), tile = 128 rows x 128 cols, K = 128.
__global__ __launch_bounds__(256) void gemm_x4_kernel(
    const float* __restrict__ X, int nrows,
    const unsigned short* __restrict__ WtQ, const unsigned short* __restrict__ WtK,
    const unsigned short* __restrict__ WtV, const unsigned short* __restrict__ WtS,
    const float* __restrict__ bq, const float* __restrict__ bk,
    const float* __restrict__ bv, const float* __restrict__ bs,
    float* __restrict__ Q, float* __restrict__ K,
    float* __restrict__ V, float* __restrict__ AGG) {
  __shared__ unsigned short ldsX[128 * LSTRIDE];  // A: rows x K, bf16
  __shared__ unsigned short ldsW[128 * LSTRIDE];  // B: [n][k] transposed, bf16

  const int r0 = blockIdx.x * 128;
  const int wave = threadIdx.x >> 5, lane = threadIdx.x & 31;
  const int lmod = lane & 15, khalf = lane >> 4;

  // stage X tile -> LDS as bf16 (zero-fill OOB rows)
  for (int idx = threadIdx.x; idx < 128 * 32; idx += 256) {
    int row = idx >> 5, ch = idx & 31;
    int gr = r0 + row;
    float4 f = (gr < nrows) ? ((const float4*)(X + (size_t)gr * HIDDIM))[ch]
                            : make_float4(0.f, 0.f, 0.f, 0.f);
    uint2 p;
    p.x = (unsigned)f2bf(f.x) | ((unsigned)f2bf(f.y) << 16);
    p.y = (unsigned)f2bf(f.z) | ((unsigned)f2bf(f.w) << 16);
    *(uint2*)&ldsX[row * LSTRIDE + ch * 4] = p;
  }
  __syncthreads();

  // preload this wave's A fragments for all 4 K-steps (layout: ISA 7.12.2)
  v16bf afrag[4];
  const unsigned short* arow = &ldsX[(wave * 16 + lmod) * LSTRIDE];
  #pragma unroll
  for (int ks = 0; ks < 4; ++ks) {
    int kb = ks * 32 + khalf * 8;
    FragU au;
    au.q[0] = *(const uint4*)(arow + kb);
    au.q[1] = *(const uint4*)(arow + kb + 16);
    afrag[ks] = au.v;
  }

  const unsigned short* wmats[4]  = {WtQ, WtK, WtV, WtS};
  const float*          biases[4] = {bq, bk, bv, bs};
  float*                outs[4]   = {Q, K, V, AGG};

  for (int wm = 0; wm < 4; ++wm) {
    __syncthreads();  // previous matrix's compute done before overwrite
    const unsigned short* wsrc = wmats[wm];
    for (int idx = threadIdx.x; idx < 128 * 16; idx += 256) {  // 16B chunks
      int rn = idx >> 4, ch = idx & 15;
      *(uint4*)&ldsW[rn * LSTRIDE + ch * 8] = *(const uint4*)&wsrc[rn * 128 + ch * 8];
    }
    __syncthreads();

    v8f acc[8];
    #pragma unroll
    for (int nt = 0; nt < 8; ++nt) { v8f z = {0,0,0,0,0,0,0,0}; acc[nt] = z; }

    #pragma unroll
    for (int nt = 0; nt < 8; ++nt) {
      const unsigned short* brow = &ldsW[(nt * 16 + lmod) * LSTRIDE];
      #pragma unroll
      for (int ks = 0; ks < 4; ++ks) {
        int kb = ks * 32 + khalf * 8;
        FragU bu;
        bu.q[0] = *(const uint4*)(brow + kb);
        bu.q[1] = *(const uint4*)(brow + kb + 16);
        acc[nt] = __builtin_amdgcn_wmma_f32_16x16x32_bf16(
            false, afrag[ks], false, bu.v, (short)0, acc[nt], false, false);
      }
    }

    float* out = outs[wm];
    const float* bias = biases[wm];
    #pragma unroll
    for (int nt = 0; nt < 8; ++nt) {
      int col = nt * 16 + lmod;
      float bb = bias[col];
      #pragma unroll
      for (int r = 0; r < 8; ++r) {
        int grow = r0 + wave * 16 + khalf * 8 + r;
        if (grow < nrows) out[(size_t)grow * HIDDIM + col] = acc[nt][r] + bb;
      }
    }
  }
}

// ---------------- edge phase 1: logits + segment max ----------------------
__global__ __launch_bounds__(256) void edge_alpha_kernel(
    const int* __restrict__ srcI, const int* __restrict__ dstI,
    const float* __restrict__ Q, const float* __restrict__ K,
    float* __restrict__ alpha, float* __restrict__ amax, int nedges) {
  int e = blockIdx.x * 256 + threadIdx.x;
  if (e >= nedges) return;
  int s = srcI[e], d = dstI[e];
  const float4* qr = (const float4*)(Q + (size_t)d * HIDDIM);
  const float4* kr = (const float4*)(K + (size_t)s * HIDDIM);
  #pragma unroll
  for (int h = 0; h < HEADS; ++h) {
    float a = 0.f;
    #pragma unroll
    for (int c = 0; c < 8; ++c) {
      float4 q4 = qr[h * 8 + c], k4 = kr[h * 8 + c];
      a += q4.x * k4.x + q4.y * k4.y + q4.z * k4.z + q4.w * k4.w;
    }
    a *= 0.17677669529663687f;  // 1/sqrt(32)
    alpha[(size_t)e * HEADS + h] = a;
    atomicMaxF(&amax[(size_t)d * HEADS + h], a);
  }
}

// ---------------- edge phase 2: exp + segment sum -------------------------
__global__ __launch_bounds__(256) void edge_exp_kernel(
    const int* __restrict__ dstI, float* __restrict__ alpha,
    const float* __restrict__ amax, float* __restrict__ denom, int nedges) {
  int e = blockIdx.x * 256 + threadIdx.x;
  if (e >= nedges) return;
  int d = dstI[e];
  #pragma unroll
  for (int h = 0; h < HEADS; ++h) {
    float ex = __expf(alpha[(size_t)e * HEADS + h] - amax[(size_t)d * HEADS + h]);
    alpha[(size_t)e * HEADS + h] = ex;
    unsafeAtomicAdd(&denom[(size_t)d * HEADS + h], ex);  // global_atomic_add_f32
  }
}

// ---------------- edge phase 3: weighted message scatter-add --------------
__global__ __launch_bounds__(256) void edge_msg_kernel(
    const int* __restrict__ srcI, const int* __restrict__ dstI,
    const float* __restrict__ V, const float* __restrict__ alpha,
    const float* __restrict__ denom, float* __restrict__ AGG, int nedges) {
  int e = blockIdx.x * 256 + threadIdx.x;
  if (e >= nedges) return;
  int s = srcI[e], d = dstI[e];
  float w[HEADS];
  #pragma unroll
  for (int h = 0; h < HEADS; ++h)
    w[h] = alpha[(size_t)e * HEADS + h] /
           (denom[(size_t)d * HEADS + h] + 1e-16f);
  const float4* vr = (const float4*)(V + (size_t)s * HIDDIM);
  float* ag = AGG + (size_t)d * HIDDIM;
  #pragma unroll
  for (int h = 0; h < HEADS; ++h) {
    #pragma unroll
    for (int c = 0; c < 8; ++c) {
      float4 v4 = vr[h * 8 + c];
      int base = h * 32 + c * 4;
      unsafeAtomicAdd(&ag[base + 0], v4.x * w[h]);
      unsafeAtomicAdd(&ag[base + 1], v4.y * w[h]);
      unsafeAtomicAdd(&ag[base + 2], v4.z * w[h]);
      unsafeAtomicAdd(&ag[base + 3], v4.w * w[h]);
    }
  }
}

// ---------------- LayerNorm + ReLU (one wave32 per row) -------------------
__global__ __launch_bounds__(256) void ln_relu_kernel(
    const float* __restrict__ AGG, const float* __restrict__ gamma,
    const float* __restrict__ beta, float* __restrict__ Hout, int nrows) {
  int wave = threadIdx.x >> 5, lane = threadIdx.x & 31;
  int row = blockIdx.x * 8 + wave;
  if (row >= nrows) return;
  float4 x = ((const float4*)(AGG + (size_t)row * HIDDIM))[lane];
  float s = x.x + x.y + x.z + x.w;
  #pragma unroll
  for (int off = 16; off > 0; off >>= 1) s += __shfl_xor(s, off, 32);
  float mu = s * (1.0f / 128.0f);
  float dx = x.x - mu, dy = x.y - mu, dz = x.z - mu, dw = x.w - mu;
  float v = dx * dx + dy * dy + dz * dz + dw * dw;
  #pragma unroll
  for (int off = 16; off > 0; off >>= 1) v += __shfl_xor(v, off, 32);
  float rs = rsqrtf(v * (1.0f / 128.0f) + 1e-5f);
  float4 g = ((const float4*)gamma)[lane];
  float4 b = ((const float4*)beta)[lane];
  float4 o;
  o.x = fmaxf(dx * rs * g.x + b.x, 0.f);
  o.y = fmaxf(dy * rs * g.y + b.y, 0.f);
  o.z = fmaxf(dz * rs * g.z + b.z, 0.f);
  o.w = fmaxf(dw * rs * g.w + b.w, 0.f);
  ((float4*)(Hout + (size_t)row * HIDDIM))[lane] = o;
}

// ---------------- classifier head: [N,128] @ [128,40] via WMMA ------------
__global__ __launch_bounds__(256) void out_gemm_kernel(
    const float* __restrict__ Hin, int nrows,
    const unsigned short* __restrict__ WtO, const float* __restrict__ bout,
    float* __restrict__ Yout) {
  __shared__ unsigned short ldsX[128 * LSTRIDE];
  __shared__ unsigned short ldsW[NCLS_PAD * LSTRIDE];

  const int r0 = blockIdx.x * 128;
  const int wave = threadIdx.x >> 5, lane = threadIdx.x & 31;
  const int lmod = lane & 15, khalf = lane >> 4;

  for (int idx = threadIdx.x; idx < 128 * 32; idx += 256) {
    int row = idx >> 5, ch = idx & 31;
    int gr = r0 + row;
    float4 f = (gr < nrows) ? ((const float4*)(Hin + (size_t)gr * HIDDIM))[ch]
                            : make_float4(0.f, 0.f, 0.f, 0.f);
    uint2 p;
    p.x = (unsigned)f2bf(f.x) | ((unsigned)f2bf(f.y) << 16);
    p.y = (unsigned)f2bf(f.z) | ((unsigned)f2bf(f.w) << 16);
    *(uint2*)&ldsX[row * LSTRIDE + ch * 4] = p;
  }
  for (int idx = threadIdx.x; idx < NCLS_PAD * 16; idx += 256) {
    int rn = idx >> 4, ch = idx & 15;
    *(uint4*)&ldsW[rn * LSTRIDE + ch * 8] = *(const uint4*)&WtO[rn * 128 + ch * 8];
  }
  __syncthreads();

  v16bf afrag[4];
  const unsigned short* arow = &ldsX[(wave * 16 + lmod) * LSTRIDE];
  #pragma unroll
  for (int ks = 0; ks < 4; ++ks) {
    int kb = ks * 32 + khalf * 8;
    FragU au;
    au.q[0] = *(const uint4*)(arow + kb);
    au.q[1] = *(const uint4*)(arow + kb + 16);
    afrag[ks] = au.v;
  }

  v8f acc[3];
  #pragma unroll
  for (int nt = 0; nt < 3; ++nt) { v8f z = {0,0,0,0,0,0,0,0}; acc[nt] = z; }
  #pragma unroll
  for (int nt = 0; nt < 3; ++nt) {
    const unsigned short* brow = &ldsW[(nt * 16 + lmod) * LSTRIDE];
    #pragma unroll
    for (int ks = 0; ks < 4; ++ks) {
      int kb = ks * 32 + khalf * 8;
      FragU bu;
      bu.q[0] = *(const uint4*)(brow + kb);
      bu.q[1] = *(const uint4*)(brow + kb + 16);
      acc[nt] = __builtin_amdgcn_wmma_f32_16x16x32_bf16(
          false, afrag[ks], false, bu.v, (short)0, acc[nt], false, false);
    }
  }

  #pragma unroll
  for (int nt = 0; nt < 3; ++nt) {
    int col = nt * 16 + lmod;
    if (col < NCLS_) {
      float bb = bout[col];
      #pragma unroll
      for (int r = 0; r < 8; ++r) {
        int grow = r0 + wave * 16 + khalf * 8 + r;
        if (grow < nrows) Yout[(size_t)grow * NCLS_ + col] = acc[nt][r] + bb;
      }
    }
  }
}

// ---------------------------------------------------------------------------
extern "C" void kernel_launch(void* const* d_in, const int* in_sizes, int n_in,
                              void* d_out, int out_size, void* d_ws, size_t ws_size,
                              hipStream_t stream) {
  const float* x     = (const float*)d_in[0];
  const int*   ei    = (const int*)  d_in[1];
  const float* Wq    = (const float*)d_in[2];
  const float* bq    = (const float*)d_in[3];
  const float* Wk    = (const float*)d_in[4];
  const float* bk    = (const float*)d_in[5];
  const float* Wv    = (const float*)d_in[6];
  const float* bv    = (const float*)d_in[7];
  const float* Ws    = (const float*)d_in[8];
  const float* bs    = (const float*)d_in[9];
  const float* gamma = (const float*)d_in[10];
  const float* beta  = (const float*)d_in[11];
  const float* Wout  = (const float*)d_in[12];
  const float* bout  = (const float*)d_in[13];

  const int N = in_sizes[0] / HIDDIM;
  const int E = in_sizes[1] / 2;
  const int* srcI = ei;
  const int* dstI = ei + E;

  // ---- workspace carve-out ----
  char* wp = (char*)d_ws;
  auto carve = [&](size_t bytes) -> char* {
    char* p = wp; wp += (bytes + 255) & ~(size_t)255; return p;
  };
  float* Q     = (float*)carve((size_t)N * HIDDIM * 4);
  float* Kb    = (float*)carve((size_t)N * HIDDIM * 4);
  float* V     = (float*)carve((size_t)N * HIDDIM * 4);
  float* AGG   = (float*)carve((size_t)N * HIDDIM * 4);
  float* Hbuf  = (float*)carve((size_t)N * HIDDIM * 4);
  float* alpha = (float*)carve((size_t)E * HEADS * 4);
  float* amax  = (float*)carve((size_t)N * HEADS * 4);
  float* denom = (float*)carve((size_t)N * HEADS * 4);
  unsigned short* WtQ[2]; unsigned short* WtK[2];
  unsigned short* WtV[2]; unsigned short* WtS[2];
  for (int l = 0; l < 2; ++l) {
    WtQ[l] = (unsigned short*)carve(128 * 128 * 2);
    WtK[l] = (unsigned short*)carve(128 * 128 * 2);
    WtV[l] = (unsigned short*)carve(128 * 128 * 2);
    WtS[l] = (unsigned short*)carve(128 * 128 * 2);
  }
  unsigned short* WtO = (unsigned short*)carve(NCLS_PAD * 128 * 2);

  // ---- weight conversion (bf16, transposed) ----
  for (int l = 0; l < 2; ++l) {
    size_t wo = (size_t)l * 128 * 128;
    convt128_kernel<<<64, 256, 0, stream>>>(Wq + wo, WtQ[l]);
    convt128_kernel<<<64, 256, 0, stream>>>(Wk + wo, WtK[l]);
    convt128_kernel<<<64, 256, 0, stream>>>(Wv + wo, WtV[l]);
    convt128_kernel<<<64, 256, 0, stream>>>(Ws + wo, WtS[l]);
  }
  convtout_kernel<<<(NCLS_PAD * 128 + 255) / 256, 256, 0, stream>>>(Wout, WtO);

  const int gemmBlocks = (N + 127) / 128;
  const int edgeBlocks = (E + 255) / 256;
  const int attnBlocks = (N * HEADS + 255) / 256;
  const int lnBlocks   = (N + 7) / 8;

  const float* hin = x;
  for (int l = 0; l < 2; ++l) {
    init_attn_kernel<<<attnBlocks, 256, 0, stream>>>(amax, denom, N * HEADS);
    gemm_x4_kernel<<<gemmBlocks, 256, 0, stream>>>(
        hin, N, WtQ[l], WtK[l], WtV[l], WtS[l],
        bq + l * HIDDIM, bk + l * HIDDIM, bv + l * HIDDIM, bs + l * HIDDIM,
        Q, Kb, V, AGG);  // AGG initialized with skip (x@Ws+bs)
    edge_alpha_kernel<<<edgeBlocks, 256, 0, stream>>>(srcI, dstI, Q, Kb, alpha, amax, E);
    edge_exp_kernel<<<edgeBlocks, 256, 0, stream>>>(dstI, alpha, amax, denom, E);
    edge_msg_kernel<<<edgeBlocks, 256, 0, stream>>>(srcI, dstI, V, alpha, denom, AGG, E);
    ln_relu_kernel<<<lnBlocks, 256, 0, stream>>>(
        AGG, gamma + l * HIDDIM, beta + l * HIDDIM, Hbuf, N);
    hin = Hbuf;
  }
  out_gemm_kernel<<<gemmBlocks, 256, 0, stream>>>(Hbuf, N, WtO, bout, (float*)d_out);
}
